// TwoPathDGCNN_19456201851244
// MI455X (gfx1250) — compile-verified, matrix operand order checked by reading, not run on hardware
//
#include <hip/hip_runtime.h>
#include <hip/hip_bf16.h>
#include <cmath>

// ---------------------------------------------------------------------------
// TwoPathDGCNN forward for gfx1250 (MI455X, wave32, WMMA).
// GEMM-heavy stages (edge 1x1 convs, attention projections, QK^T, A*V) run
// on v_wmma_f32_16x16x32_bf16 with 2x2 output tiles per wave (32x32/wave,
// 64x64 per 4-wave block): f32 global data is staged into LDS as RNE-rounded
// bf16 in WMMA fragment order, 4 WMMA issues per barrier round-trip, f32
// accumulation held strictly in VGPRs (no address-taken accumulators -> no
// scratch spills). Reductions (kNN top-k, BN stats, softmax, pooling,
// decoders) are wave32 VALU kernels with LDS tree reductions.
// ---------------------------------------------------------------------------

#define BATCH 8
#define NPTS  1024
#define KNN   20

typedef __attribute__((ext_vector_type(16))) __bf16          v16bf;
typedef __attribute__((ext_vector_type(16))) unsigned short  v16us;
typedef __attribute__((ext_vector_type(8)))  float           v8f;

// float -> bf16, round-to-nearest-even (pure integer; no __bf16 arithmetic)
__device__ __forceinline__ unsigned short f2bf(float f) {
  unsigned u = __builtin_bit_cast(unsigned, f);
  u += 0x7FFFu + ((u >> 16) & 1u);
  return (unsigned short)(u >> 16);
}

// 16-bit A/B fragment K-index map (ISA 7.12.2, 16-bit A-matrix 16x32):
// lanes 0-15 : e=0..7 -> K=0..7,  e=8..15 -> K=16..23
// lanes 16-31: e=0..7 -> K=8..15, e=8..15 -> K=24..31
__device__ __forceinline__ int kmap(int lane, int e) {
  return ((e < 8) ? e : (e + 8)) + ((lane >> 4) << 3);
}

__device__ __forceinline__ v16bf ldfrag(const unsigned short* p) {
  return __builtin_bit_cast(v16bf, *(const v16us*)p);
}

__device__ __forceinline__ v8f wmma_bf16(v16bf a, v16bf b, v8f c) {
  return __builtin_amdgcn_wmma_f32_16x16x32_bf16(false, a, false, b, (short)0, c,
                                                 false, false);
}

// C/D tile store helpers: acc passed BY VALUE so accumulators are never
// address-taken (keeps them in VGPRs; avoids scratch spill/fill in K-loop).
__device__ __forceinline__ void store8(float* __restrict__ p, size_t stride,
                                       v8f acc, float s) {
#pragma unroll
  for (int r = 0; r < 8; ++r) p[(size_t)r * stride] = acc[r] * s;
}
__device__ __forceinline__ void store8b(float* __restrict__ p, size_t stride,
                                        v8f acc, const float* bias8) {
#pragma unroll
  for (int r = 0; r < 8; ++r) p[(size_t)r * stride] = acc[r] + (bias8 ? bias8[r] : 0.f);
}

__device__ __forceinline__ float gelu_exact(float v) {
  return 0.5f * v * (1.f + erff(v * 0.70710678118654752f));
}

// ---------------------------------------------------------------------------
// Slice path p out of x[B,2,3,N] -> xp[B,3,N]
// ---------------------------------------------------------------------------
__global__ void slice_path(const float* __restrict__ x, int p, float* __restrict__ out) {
  size_t i = (size_t)blockIdx.x * blockDim.x + threadIdx.x;
  if (i >= (size_t)BATCH * 3 * NPTS) return;
  int n = i % NPTS;
  int c = (i / NPTS) % 3;
  int b = i / (3 * NPTS);
  out[i] = x[((((size_t)b * 2 + p) * 3) + c) * NPTS + n];
}

// ---------------------------------------------------------------------------
// kNN: per (b,n) row, top-20 by (2*dot(q,m) - |m|^2); the -|q|^2 term is
// constant per row and dropped (ordering-invariant). Self is rank 0,
// matching lax.top_k on the reference distance matrix.
// ---------------------------------------------------------------------------
__global__ __launch_bounds__(256) void knn_kernel(const float* __restrict__ x, int C,
                                                  int* __restrict__ idx) {
  int b = blockIdx.x / NPTS;
  int n = blockIdx.x % NPTS;
  const float* xb = x + (size_t)b * C * NPTS;
  __shared__ float dist[NPTS];
  __shared__ float q[256];
  __shared__ float rv[256];
  __shared__ int   ri[256];
  int t = threadIdx.x;

  for (int c = t; c < C; c += 256) q[c] = xb[(size_t)c * NPTS + n];
  __syncthreads();

  for (int m = t; m < NPTS; m += 256) {
    float dot = 0.f, mm = 0.f;
    for (int c = 0; c < C; ++c) {
      float v = xb[(size_t)c * NPTS + m];
      dot += v * q[c];
      mm  += v * v;
    }
    dist[m] = 2.f * dot - mm;
  }
  __syncthreads();

  for (int s = 0; s < KNN; ++s) {
    float bv = -3.0e38f; int bi = 0;
    for (int m = t; m < NPTS; m += 256) {
      float v = dist[m];
      if (v > bv || (v == bv && m < bi)) { bv = v; bi = m; }
    }
    rv[t] = bv; ri[t] = bi;
    __syncthreads();
    for (int w = 128; w > 0; w >>= 1) {
      if (t < w) {
        float ov = rv[t + w]; int oi = ri[t + w];
        if (ov > rv[t] || (ov == rv[t] && oi < ri[t])) { rv[t] = ov; ri[t] = oi; }
      }
      __syncthreads();
    }
    if (t == 0) {
      idx[((size_t)b * NPTS + n) * KNN + s] = ri[0];
      dist[ri[0]] = -3.0e38f;
    }
    __syncthreads();
  }
}

// ---------------------------------------------------------------------------
// Edge-conv GEMM: Y[o,j] = sum_c W[o,c] * feat(c,j), j over B*N*KNN columns,
// feat(c<Cin) = x[neighbor]-x[center], feat(Cin..2Cin) = x[center].
// Block = 128 threads = 4 waves in a 2x2 grid; each wave owns a 32x32 tile
// (2x2 WMMA accumulators) of the 64x64 block tile.
// ---------------------------------------------------------------------------
__global__ __launch_bounds__(128) void edge_gemm(const float* __restrict__ W,  // [Cout, 2Cin]
                                                 const float* __restrict__ X,  // [B, Cin, N]
                                                 const int*   __restrict__ idx,// [B, N, KNN]
                                                 float* __restrict__ Y,        // [Cout, B*N*KNN]
                                                 int Cin, int Cout) {
  const int cols = BATCH * NPTS * KNN;
  const int ct = blockIdx.x * 64;   // column tile
  const int mt = blockIdx.y * 64;   // output-channel tile
  const int t = threadIdx.x;
  const int lane = t & 31, wave = t >> 5;
  const int wm = wave >> 1, wn = wave & 1;   // 2x2 wave grid

  __shared__ __align__(32) unsigned short Af[4][32][16];
  __shared__ __align__(32) unsigned short Bf[4][32][16];
  __shared__ int cB[64], cN[64], cI[64];

  if (t < 64) {
    int j = ct + t;
    int b = j / (NPTS * KNN);
    int r = j % (NPTS * KNN);
    cB[t] = b;
    cN[t] = r / KNN;
    cI[t] = idx[((size_t)b * NPTS + (r / KNN)) * KNN + (r % KNN)];
  }
  __syncthreads();

  const int K2 = 2 * Cin;
  v8f a00 = {}, a01 = {}, a10 = {}, a11 = {};
  for (int kb = 0; kb < K2; kb += 32) {
    if (kb + 32 < K2)  // prefetch next weight K-tile -> global_prefetch_b8
      __builtin_prefetch(&W[(size_t)(mt + (t & 63)) * K2 + kb + 32], 0, 0);
    for (int i = t; i < 2048; i += 128) {
      int half = i >> 9;          // which 16-wide slice of the 64 tile
      int ln   = (i >> 4) & 31;   // fragment lane
      int e    = i & 15;          // fragment element
      int kk   = kb + kmap(ln, e);
      // A (weights)
      int row = mt + half * 16 + (ln & 15);
      Af[half][ln][e] = f2bf((kk < K2) ? W[(size_t)row * K2 + kk] : 0.f);
      // B (edge features, gathered + differenced on the fly)
      int col = half * 16 + (ln & 15);
      int b = cB[col], n = cN[col], id = cI[col];
      float v;
      if (kk < Cin)
        v = X[((size_t)b * Cin + kk) * NPTS + id] - X[((size_t)b * Cin + kk) * NPTS + n];
      else if (kk < K2)
        v = X[((size_t)b * Cin + (kk - Cin)) * NPTS + n];
      else
        v = 0.f;
      Bf[half][ln][e] = f2bf(v);
    }
    __syncthreads();
    v16bf fa0 = ldfrag(&Af[wm * 2 + 0][lane][0]);
    v16bf fa1 = ldfrag(&Af[wm * 2 + 1][lane][0]);
    v16bf fb0 = ldfrag(&Bf[wn * 2 + 0][lane][0]);
    v16bf fb1 = ldfrag(&Bf[wn * 2 + 1][lane][0]);
    a00 = wmma_bf16(fa0, fb0, a00);
    a01 = wmma_bf16(fa0, fb1, a01);
    a10 = wmma_bf16(fa1, fb0, a10);
    a11 = wmma_bf16(fa1, fb1, a11);
    __syncthreads();
  }

  const int rb = (lane >> 4) << 3;            // C/D layout row offset
  const int r0 = mt + wm * 32 + rb;
  const int c0 = ct + wn * 32 + (lane & 15);
  store8(&Y[(size_t)r0 * cols + c0],             cols, a00, 1.f);
  store8(&Y[(size_t)r0 * cols + c0 + 16],        cols, a01, 1.f);
  store8(&Y[(size_t)(r0 + 16) * cols + c0],      cols, a10, 1.f);
  store8(&Y[(size_t)(r0 + 16) * cols + c0 + 16], cols, a11, 1.f);
}

// ---------------------------------------------------------------------------
// Per-channel batch stats over all columns -> BN scale/shift (biased var).
// ---------------------------------------------------------------------------
__global__ __launch_bounds__(256) void colstats(const float* __restrict__ Y, int cols,
                                                const float* __restrict__ g,
                                                const float* __restrict__ bt,
                                                float* __restrict__ scale,
                                                float* __restrict__ shift) {
  int o = blockIdx.x;
  const float* row = Y + (size_t)o * cols;
  int t = threadIdx.x;
  float s = 0.f, s2 = 0.f;
  for (int j = t; j < cols; j += 256) { float v = row[j]; s += v; s2 += v * v; }
  __shared__ float rs[256], rq[256];
  rs[t] = s; rq[t] = s2;
  __syncthreads();
  for (int w = 128; w > 0; w >>= 1) {
    if (t < w) { rs[t] += rs[t + w]; rq[t] += rq[t + w]; }
    __syncthreads();
  }
  if (t == 0) {
    float m   = rs[0] / (float)cols;
    float var = rq[0] / (float)cols - m * m;
    float sc  = g[o] * rsqrtf(var + 1e-5f);
    scale[o] = sc;
    shift[o] = bt[o] - m * sc;
  }
}

// ---------------------------------------------------------------------------
// Apply BN + exact GELU, then max over the KNN axis -> act[B,Cout,N]
// ---------------------------------------------------------------------------
__global__ __launch_bounds__(256) void bn_gelu_maxk(const float* __restrict__ Y,
                                                    const float* __restrict__ scale,
                                                    const float* __restrict__ shift,
                                                    float* __restrict__ out, int Cout) {
  const int cols = BATCH * NPTS * KNN;
  size_t tid = (size_t)blockIdx.x * blockDim.x + threadIdx.x;
  size_t total = (size_t)BATCH * Cout * NPTS;
  if (tid >= total) return;
  int n = tid % NPTS;
  int o = (tid / NPTS) % Cout;
  int b = tid / ((size_t)NPTS * Cout);
  size_t jb = ((size_t)b * NPTS + n) * KNN;
  float sc = scale[o], sh = shift[o], m = -3.0e38f;
  for (int kk = 0; kk < KNN; ++kk) {
    float v = Y[(size_t)o * cols + jb + kk] * sc + sh;
    m = fmaxf(m, gelu_exact(v));
  }
  out[((size_t)b * Cout + o) * NPTS + n] = m;
}

// ---------------------------------------------------------------------------
// Generic WMMA GEMM over point columns: Y[b,o,n] = sum_c W[o,c]*X[b,c,n]+bias
// 64x64 block tile, 2x2 WMMA per wave. (attention Q/K/V and out projections)
// ---------------------------------------------------------------------------
__global__ __launch_bounds__(128) void gemm_bcn(const float* __restrict__ W,    // [Cout,Cin]
                                                const float* __restrict__ bias, // [Cout] or null
                                                const float* __restrict__ X,    // [B,Cin,N]
                                                float* __restrict__ Y,          // [B,Cout,N]
                                                int Cin, int Cout) {
  const int ct = blockIdx.x * 64;  // over B*N
  const int mt = blockIdx.y * 64;
  const int t = threadIdx.x;
  const int lane = t & 31, wave = t >> 5;
  const int wm = wave >> 1, wn = wave & 1;
  __shared__ __align__(32) unsigned short Af[4][32][16];
  __shared__ __align__(32) unsigned short Bf[4][32][16];

  v8f a00 = {}, a01 = {}, a10 = {}, a11 = {};
  for (int kb = 0; kb < Cin; kb += 32) {
    if (kb + 32 < Cin)
      __builtin_prefetch(&W[(size_t)(mt + (t & 63)) * Cin + kb + 32], 0, 0);
    for (int i = t; i < 2048; i += 128) {
      int half = i >> 9, ln = (i >> 4) & 31, e = i & 15;
      int kk = kb + kmap(ln, e);
      int row = mt + half * 16 + (ln & 15);
      Af[half][ln][e] = f2bf((kk < Cin) ? W[(size_t)row * Cin + kk] : 0.f);
      int j = ct + half * 16 + (ln & 15);
      int b = j / NPTS, n = j % NPTS;
      Bf[half][ln][e] = f2bf((kk < Cin) ? X[((size_t)b * Cin + kk) * NPTS + n] : 0.f);
    }
    __syncthreads();
    v16bf fa0 = ldfrag(&Af[wm * 2 + 0][lane][0]);
    v16bf fa1 = ldfrag(&Af[wm * 2 + 1][lane][0]);
    v16bf fb0 = ldfrag(&Bf[wn * 2 + 0][lane][0]);
    v16bf fb1 = ldfrag(&Bf[wn * 2 + 1][lane][0]);
    a00 = wmma_bf16(fa0, fb0, a00);
    a01 = wmma_bf16(fa0, fb1, a01);
    a10 = wmma_bf16(fa1, fb0, a10);
    a11 = wmma_bf16(fa1, fb1, a11);
    __syncthreads();
  }
  const int rb = (lane >> 4) << 3;
  const int o0 = mt + wm * 32 + rb;                 // output-channel base (tile mi=0)
  const int j0 = ct + wn * 32 + (lane & 15);        // point column (tile ni=0)
  const int b  = j0 / NPTS, n = j0 % NPTS;          // 64-tiles never straddle batches
  float* base = Y + ((size_t)b * Cout + o0) * NPTS + n;
  store8b(base,                          NPTS, a00, bias ? bias + o0 : nullptr);
  store8b(base + 16,                     NPTS, a01, bias ? bias + o0 : nullptr);
  store8b(base + (size_t)16 * NPTS,      NPTS, a10, bias ? bias + o0 + 16 : nullptr);
  store8b(base + (size_t)16 * NPTS + 16, NPTS, a11, bias ? bias + o0 + 16 : nullptr);
}

// ---------------------------------------------------------------------------
// Attention scores: S[b,h,n,m] = (1/sqrt(C)) * sum_d Q[b,h*32+d,n]*K[b,h*32+d,m]
// Head dim is 32 for every block in this model -> single WMMA K-step.
// 64x64 block tile over (n,m), 2x2 WMMA per wave.
// ---------------------------------------------------------------------------
__global__ __launch_bounds__(128) void attn_scores(const float* __restrict__ Q,
                                                   const float* __restrict__ Kt,
                                                   float* __restrict__ S,
                                                   int C, int H, float invs) {
  const int bh = blockIdx.z, b = bh / H, h = bh % H;
  const int ntile = blockIdx.y * 64, mtile = blockIdx.x * 64;
  const int t = threadIdx.x;
  const int lane = t & 31, wave = t >> 5;
  const int wm = wave >> 1, wn = wave & 1;
  __shared__ __align__(32) unsigned short Af[4][32][16];
  __shared__ __align__(32) unsigned short Bf[4][32][16];

  for (int i = t; i < 2048; i += 128) {
    int half = i >> 9, ln = (i >> 4) & 31, e = i & 15;
    int d = kmap(ln, e);
    Af[half][ln][e] = f2bf(Q[((size_t)b * C + h * 32 + d) * NPTS + ntile + half * 16 + (ln & 15)]);
    Bf[half][ln][e] = f2bf(Kt[((size_t)b * C + h * 32 + d) * NPTS + mtile + half * 16 + (ln & 15)]);
  }
  __syncthreads();
  v16bf fa0 = ldfrag(&Af[wm * 2 + 0][lane][0]);
  v16bf fa1 = ldfrag(&Af[wm * 2 + 1][lane][0]);
  v16bf fb0 = ldfrag(&Bf[wn * 2 + 0][lane][0]);
  v16bf fb1 = ldfrag(&Bf[wn * 2 + 1][lane][0]);
  v8f a00 = {}, a01 = {}, a10 = {}, a11 = {};
  a00 = wmma_bf16(fa0, fb0, a00);
  a01 = wmma_bf16(fa0, fb1, a01);
  a10 = wmma_bf16(fa1, fb0, a10);
  a11 = wmma_bf16(fa1, fb1, a11);

  const int rb = (lane >> 4) << 3;
  const int n0 = ntile + wm * 32 + rb;
  const int m0 = mtile + wn * 32 + (lane & 15);
  float* base = S + ((size_t)bh * NPTS + n0) * NPTS + m0;
  store8(base,                          NPTS, a00, invs);
  store8(base + 16,                     NPTS, a01, invs);
  store8(base + (size_t)16 * NPTS,      NPTS, a10, invs);
  store8(base + (size_t)16 * NPTS + 16, NPTS, a11, invs);
}

__global__ __launch_bounds__(256) void softmax_rows(float* __restrict__ S) {
  float* p = S + (size_t)blockIdx.x * NPTS;
  int t = threadIdx.x;
  __shared__ float red[256];
  float mx = -3.0e38f;
  for (int m = t; m < NPTS; m += 256) mx = fmaxf(mx, p[m]);
  red[t] = mx; __syncthreads();
  for (int w = 128; w > 0; w >>= 1) { if (t < w) red[t] = fmaxf(red[t], red[t + w]); __syncthreads(); }
  mx = red[0]; __syncthreads();
  float s = 0.f;
  for (int m = t; m < NPTS; m += 256) { float e = __expf(p[m] - mx); p[m] = e; s += e; }
  red[t] = s; __syncthreads();
  for (int w = 128; w > 0; w >>= 1) { if (t < w) red[t] += red[t + w]; __syncthreads(); }
  float inv = 1.f / red[0];
  for (int m = t; m < NPTS; m += 256) p[m] *= inv;
}

// ---------------------------------------------------------------------------
// ctx[b, h*32+d, n] = sum_m P[b,h,n,m] * V[b, h*32+d, m]  (K loop over 1024)
// 32 rows (one head) x 128 cols per block: 4 waves side-by-side in N, each
// 32x32 via 2x2 WMMA; V fragments shared through LDS across all 4 waves.
// ---------------------------------------------------------------------------
__global__ __launch_bounds__(128) void attn_av(const float* __restrict__ P,
                                               const float* __restrict__ V,
                                               float* __restrict__ ctx, int C, int H) {
  const int bh = blockIdx.z, b = bh / H, h = bh % H;
  const int nt = blockIdx.x * 128;
  const int t = threadIdx.x;
  const int lane = t & 31, wave = t >> 5;   // wave = column sub-tile 0..3
  __shared__ __align__(32) unsigned short Af[2][32][16];
  __shared__ __align__(32) unsigned short Bf[8][32][16];

  v8f a00 = {}, a01 = {}, a10 = {}, a11 = {};
  for (int kb = 0; kb < NPTS; kb += 32) {
    for (int i = t; i < 1024; i += 128) {       // stage V (A-matrix, 32 rows)
      int half = i >> 9, ln = (i >> 4) & 31, e = i & 15;
      int m = kb + kmap(ln, e);
      Af[half][ln][e] = f2bf(V[((size_t)b * C + h * 32 + half * 16 + (ln & 15)) * NPTS + m]);
    }
    for (int i = t; i < 4096; i += 128) {       // stage P rows (B-matrix, 128 cols)
      int half = i >> 9, ln = (i >> 4) & 31, e = i & 15;
      int m = kb + kmap(ln, e);
      Bf[half][ln][e] = f2bf(P[(((size_t)bh * NPTS) + nt + half * 16 + (ln & 15)) * NPTS + m]);
    }
    __syncthreads();
    v16bf fa0 = ldfrag(&Af[0][lane][0]);
    v16bf fa1 = ldfrag(&Af[1][lane][0]);
    v16bf fb0 = ldfrag(&Bf[wave * 2 + 0][lane][0]);
    v16bf fb1 = ldfrag(&Bf[wave * 2 + 1][lane][0]);
    a00 = wmma_bf16(fa0, fb0, a00);
    a01 = wmma_bf16(fa0, fb1, a01);
    a10 = wmma_bf16(fa1, fb0, a10);
    a11 = wmma_bf16(fa1, fb1, a11);
    __syncthreads();
  }
  const int rb = (lane >> 4) << 3;
  const int d0 = rb;                                // head-row base (tile mi=0)
  const int n0 = nt + wave * 32 + (lane & 15);
  float* base = ctx + ((size_t)b * C + h * 32 + d0) * NPTS + n0;
  store8(base,                          NPTS, a00, 1.f);
  store8(base + 16,                     NPTS, a01, 1.f);
  store8(base + (size_t)16 * NPTS,      NPTS, a10, 1.f);
  store8(base + (size_t)16 * NPTS + 16, NPTS, a11, 1.f);
}

// ---------------------------------------------------------------------------
// pools(concat(f1,f2)) : out[b, i] = max_n, out[b, 2C+i] = mean_n, i in [0,2C)
// ---------------------------------------------------------------------------
__global__ __launch_bounds__(256) void pool2(const float* __restrict__ f1,
                                             const float* __restrict__ f2, int C,
                                             float* __restrict__ out, int outStride) {
  int b = blockIdx.x / (2 * C);
  int i = blockIdx.x % (2 * C);
  const float* src = (i < C) ? (f1 + ((size_t)b * C + i) * NPTS)
                             : (f2 + ((size_t)b * C + (i - C)) * NPTS);
  int t = threadIdx.x;
  float mx = -3.0e38f, s = 0.f;
  for (int n = t; n < NPTS; n += 256) { float v = src[n]; mx = fmaxf(mx, v); s += v; }
  __shared__ float rm[256], rs[256];
  rm[t] = mx; rs[t] = s; __syncthreads();
  for (int w = 128; w > 0; w >>= 1) {
    if (t < w) { rm[t] = fmaxf(rm[t], rm[t + w]); rs[t] += rs[t + w]; }
    __syncthreads();
  }
  if (t == 0) {
    out[(size_t)b * outStride + i]         = rm[0];
    out[(size_t)b * outStride + 2 * C + i] = rs[0] / (float)NPTS;
  }
}

// ---------------------------------------------------------------------------
// Decoder: y[b,o] = W[o,:]·in[b,:]  (B=8, F<=1024 -> plain reduction kernel)
// ---------------------------------------------------------------------------
__global__ __launch_bounds__(256) void dec_matvec(const float* __restrict__ W,
                                                  const float* __restrict__ in, int F,
                                                  float* __restrict__ y) {
  int o = blockIdx.x, t = threadIdx.x;
  __shared__ float red[256];
  for (int b = 0; b < BATCH; ++b) {
    float s = 0.f;
    for (int f = t; f < F; f += 256) s += W[(size_t)o * F + f] * in[(size_t)b * F + f];
    red[t] = s; __syncthreads();
    for (int w = 128; w > 0; w >>= 1) { if (t < w) red[t] += red[t + w]; __syncthreads(); }
    if (t == 0) y[b * 256 + o] = red[0];
    __syncthreads();
  }
}

__global__ void dec_bn_gelu(const float* __restrict__ y, const float* __restrict__ g,
                            const float* __restrict__ bt, float* __restrict__ out,
                            int outStride) {
  int o = threadIdx.x;  // 256 threads
  float s = 0.f, s2 = 0.f;
  for (int b = 0; b < BATCH; ++b) { float v = y[b * 256 + o]; s += v; s2 += v * v; }
  float m = s / BATCH, var = s2 / BATCH - m * m;
  float sc = g[o] * rsqrtf(var + 1e-5f), sh = bt[o] - m * sc;
  for (int b = 0; b < BATCH; ++b)
    out[(size_t)b * outStride + o] = gelu_exact(y[b * 256 + o] * sc + sh);
}

// ---------------------------------------------------------------------------
// Host orchestration
// ---------------------------------------------------------------------------
// Param indices: harness flattens the params dict in sorted-key (jax tree)
// order after x. ca: bk,bo,bq,bv,wk,wo,wq,wv ; dec/ec: b,g,w
enum {
  IN_X = 0,
  CA1 = 1, CA2 = 9, CA3 = 17, CA4 = 25,
  D1 = 33, D2 = 36, D3 = 39, D4 = 42,
  P1E1 = 45, P1E2 = 48, P1E3 = 51, P1E4 = 54,
  P2E1 = 57, P2E2 = 60, P2E3 = 63, P2E4 = 66
};

extern "C" void kernel_launch(void* const* d_in, const int* in_sizes, int n_in,
                              void* d_out, int out_size, void* d_ws, size_t ws_size,
                              hipStream_t stream) {
  (void)in_sizes; (void)n_in; (void)out_size; (void)ws_size;
  const float* x = (const float*)d_in[IN_X];
  auto P = [&](int i) { return (const float*)d_in[i]; };

  // ---- workspace bump allocator (peak ~354 MB) ----
  float* wsf = (float*)d_ws;
  size_t off = 0;
  auto alloc = [&](size_t n) { float* p = wsf + off; off += (n + 63) & ~(size_t)63; return p; };

  int* idxA = (int*)alloc((size_t)BATCH * NPTS * KNN);
  int* idxB = (int*)alloc((size_t)BATCH * NPTS * KNN);
  float* x1 = alloc((size_t)BATCH * 3 * NPTS);
  float* x2 = alloc((size_t)BATCH * 3 * NPTS);
  float* p1x1 = alloc((size_t)BATCH * 64 * NPTS);
  float* p2x1 = alloc((size_t)BATCH * 64 * NPTS);
  float* c1   = alloc((size_t)BATCH * 64 * NPTS);
  float* p1x2 = alloc((size_t)BATCH * 64 * NPTS);
  float* p2x2 = alloc((size_t)BATCH * 64 * NPTS);
  float* c2   = alloc((size_t)BATCH * 64 * NPTS);
  float* p1x3 = alloc((size_t)BATCH * 128 * NPTS);
  float* p2x3 = alloc((size_t)BATCH * 128 * NPTS);
  float* c3   = alloc((size_t)BATCH * 128 * NPTS);
  float* p1x4 = alloc((size_t)BATCH * 256 * NPTS);
  float* p2x4 = alloc((size_t)BATCH * 256 * NPTS);
  float* c4   = alloc((size_t)BATCH * 256 * NPTS);
  float* Qb   = alloc((size_t)BATCH * 256 * NPTS);
  float* Kb   = alloc((size_t)BATCH * 256 * NPTS);
  float* Vb   = alloc((size_t)BATCH * 256 * NPTS);
  float* ctx  = alloc((size_t)BATCH * 256 * NPTS);
  float* scale = alloc(256);
  float* shift = alloc(256);
  float* decin1 = alloc((size_t)BATCH * 1024);
  float* decin2 = alloc((size_t)BATCH * 768);
  float* decin3 = alloc((size_t)BATCH * 512);
  float* decin4 = alloc((size_t)BATCH * 512);
  float* ydec   = alloc((size_t)BATCH * 256);
  // big scratch, shared (disjoint lifetimes): edge-conv pre-act [256 x 163840]
  // vs. attention scores [B*H x 1024 x 1024] (ca4: 64M floats)
  float* big = alloc((size_t)64 * 1024 * 1024 + 1024);

  const int cols = BATCH * NPTS * KNN;

  auto run_knn = [&](const float* xin, int C, int* idxbuf) {
    knn_kernel<<<BATCH * NPTS, 256, 0, stream>>>(xin, C, idxbuf);
  };
  auto run_conv = [&](const float* Xin, int Cin, int Cout, int ecBase,
                      const int* idxbuf, float* outAct) {
    const float* eb = P(ecBase + 0);
    const float* eg = P(ecBase + 1);
    const float* ew = P(ecBase + 2);
    dim3 grid(cols / 64, Cout / 64);
    edge_gemm<<<grid, 128, 0, stream>>>(ew, Xin, idxbuf, big, Cin, Cout);
    colstats<<<Cout, 256, 0, stream>>>(big, cols, eg, eb, scale, shift);
    size_t tot = (size_t)BATCH * Cout * NPTS;
    bn_gelu_maxk<<<(unsigned)((tot + 255) / 256), 256, 0, stream>>>(big, scale, shift,
                                                                    outAct, Cout);
  };
  auto run_attn = [&](const float* q, const float* kv, int C, int H, int caBase,
                      float* outC) {
    const float* bk = P(caBase + 0); const float* bo = P(caBase + 1);
    const float* bq = P(caBase + 2); const float* bv = P(caBase + 3);
    const float* wk = P(caBase + 4); const float* wo = P(caBase + 5);
    const float* wq = P(caBase + 6); const float* wv = P(caBase + 7);
    dim3 gp((BATCH * NPTS) / 64, C / 64);
    gemm_bcn<<<gp, 128, 0, stream>>>(wq, bq, q,  Qb, C, C);
    gemm_bcn<<<gp, 128, 0, stream>>>(wk, bk, kv, Kb, C, C);
    gemm_bcn<<<gp, 128, 0, stream>>>(wv, bv, kv, Vb, C, C);
    dim3 gs(NPTS / 64, NPTS / 64, BATCH * H);
    attn_scores<<<gs, 128, 0, stream>>>(Qb, Kb, big, C, H, 1.f / sqrtf((float)C));
    softmax_rows<<<BATCH * H * NPTS, 256, 0, stream>>>(big);
    dim3 ga(NPTS / 128, 1, BATCH * H);
    attn_av<<<ga, 128, 0, stream>>>(big, Vb, ctx, C, H);
    gemm_bcn<<<gp, 128, 0, stream>>>(wo, bo, ctx, outC, C, C);
  };
  auto run_dec = [&](const float* in, int F, int dBase, float* out, int outStride) {
    const float* db = P(dBase + 0);
    const float* dg = P(dBase + 1);
    const float* dw = P(dBase + 2);
    dec_matvec<<<256, 256, 0, stream>>>(dw, in, F, ydec);
    dec_bn_gelu<<<1, 256, 0, stream>>>(ydec, dg, db, out, outStride);
  };

  // ---- pipeline ----
  {
    size_t n = (size_t)BATCH * 3 * NPTS;
    slice_path<<<(unsigned)((n + 255) / 256), 256, 0, stream>>>(x, 0, x1);
    slice_path<<<(unsigned)((n + 255) / 256), 256, 0, stream>>>(x, 1, x2);
  }
  run_knn(x1, 3, idxA);  run_conv(x1, 3, 64, P1E1, idxA, p1x1);
  run_knn(x2, 3, idxB);  run_conv(x2, 3, 64, P2E1, idxB, p2x1);
  run_attn(p1x1, p2x1, 64, 2, CA1, c1);

  run_knn(p1x1, 64, idxA);  run_conv(p1x1, 64, 64, P1E2, idxA, p1x2);
  run_knn(p2x1, 64, idxB);  run_conv(p2x1, 64, 64, P2E2, idxB, p2x2);
  run_attn(p1x2, p2x2, 64, 2, CA2, c2);

  run_knn(p1x2, 64, idxA);  run_conv(p1x2, 64, 128, P1E3, idxA, p1x3);
  run_knn(p2x2, 64, idxB);  run_conv(p2x2, 64, 128, P2E3, idxB, p2x3);
  run_attn(p1x3, p2x3, 128, 4, CA3, c3);

  run_knn(p1x3, 128, idxA); run_conv(p1x3, 128, 256, P1E4, idxA, p1x4);
  run_knn(p2x3, 128, idxB); run_conv(p2x3, 128, 256, P2E4, idxB, p2x4);
  run_attn(p1x4, p2x4, 256, 8, CA4, c4);

  // decoders: fx_i written into the first 256 columns of the next dec input,
  // pools written into the remaining columns (sequential on stream).
  pool2<<<BATCH * 2 * 256, 256, 0, stream>>>(p1x4, c4, 256, decin1, 1024);
  run_dec(decin1, 1024, D1, decin2, 768);
  pool2<<<BATCH * 2 * 128, 256, 0, stream>>>(p1x3, c3, 128, decin2 + 256, 768);
  run_dec(decin2, 768, D2, decin3, 512);
  pool2<<<BATCH * 2 * 64, 256, 0, stream>>>(p1x2, c2, 64, decin3 + 256, 512);
  run_dec(decin3, 512, D3, decin4, 512);
  pool2<<<BATCH * 2 * 64, 256, 0, stream>>>(p1x1, c1, 64, decin4 + 256, 512);
  run_dec(decin4, 512, D4, (float*)d_out, 256);
}
